// CodegenGPTJStyleAttention_3685081940088
// MI455X (gfx1250) — matrix operand order repaired and checked by reading, not compiled, not run on hardware
//
#include <hip/hip_runtime.h>

// ---------------------------------------------------------------------------
// GPT-J attention for MI455X (gfx1250, wave32, WMMA + TDM).
// All matmuls: v_wmma_f32_16x16x32_bf16 (f32 accumulate). Softmax/RoPE in f32.
// K/V tiles staged into LDS by the Tensor Data Mover (tensor_load_to_lds,
// TENSORcnt) when the builtin is available; cooperative b128 loads otherwise.
// Workspace layout (bf16, 32MB each): X, Wq, Wk, Wv, Wo, Q, K, V, Ctx = 288MB.
// ---------------------------------------------------------------------------

typedef __bf16 bf16;
typedef __attribute__((ext_vector_type(16))) __bf16 bf16x16;
typedef __attribute__((ext_vector_type(8)))  float   f32x8;

constexpr int Bc  = 2;
constexpr int Sc  = 2048;
constexpr int Hc  = 4096;
constexpr int NHc = 16;
constexpr int HDc = 256;
constexpr int RDc = 64;

#ifdef __gfx1250__
#if __has_builtin(__builtin_amdgcn_tensor_load_to_lds)
#define USE_TDM 1
#endif
#endif
#ifndef USE_TDM
#define USE_TDM 0
#endif

__device__ __forceinline__ f32x8 wmma_bf16(bf16x16 a, bf16x16 b, f32x8 c) {
  // (neg_a, A, neg_b, B, c_mod, C, reuse_a, reuse_b)
  return __builtin_amdgcn_wmma_f32_16x16x32_bf16(false, a, false, b, (short)0, c,
                                                 false, false);
}

__device__ __forceinline__ f32x8 zero8() {
  f32x8 z;
#pragma unroll
  for (int i = 0; i < 8; ++i) z[i] = 0.0f;
  return z;
}

// A fragment 16x32 (bf16), src row-major [M][K] with leading dim ld.
// lane = half*16 + m; VGPR v holds K = 2v + 16*half (contiguous, matches B).
__device__ __forceinline__ bf16x16 load_a_frag(const bf16* p, int ld, int m0,
                                               int k0, int lane) {
  const int  m    = m0 + (lane & 15);
  const int  half = lane >> 4;
  const bf16* row = p + (size_t)m * ld + k0 + 16 * half;
  bf16x16 a;
#pragma unroll
  for (int v = 0; v < 8; ++v) {
    const int kk = 2 * v;
    a[2 * v]     = row[kk];
    a[2 * v + 1] = row[kk + 1];
  }
  return a;
}

// B fragment 32x16 from a K-contiguous ("B^T") source: src[n][k], ld over k.
// Lane = half*16 + n; element e holds K = k0 + 16*half + e  -> 32B contiguous.
__device__ __forceinline__ bf16x16 load_bt_frag(const bf16* p, int ld, int k0,
                                                int n0, int lane) {
  const bf16* q = p + (size_t)(n0 + (lane & 15)) * ld + k0 + 16 * (lane >> 4);
  return *(const bf16x16*)q;
}

// B fragment 32x16 from a row-major [K][N] source (stride-ld gather over K).
__device__ __forceinline__ bf16x16 load_b_frag(const bf16* p, int ld, int k0,
                                               int n0, int lane) {
  const bf16* q = p + (size_t)(k0 + 16 * (lane >> 4)) * ld + n0 + (lane & 15);
  bf16x16 b;
#pragma unroll
  for (int e = 0; e < 16; ++e) b[e] = q[(size_t)e * ld];
  return b;
}

// ---------------------------------------------------------------------------
// TDM: async 2-D tile load (rows x HDc bf16, row-major, stride HDc) -> LDS.
// D# per ISA 8.3-8.6: group0 = {count=1 | lds_addr | global_addr | type=2},
// group1 = {data_size=2B, tensor_dim0=HDc, tensor_dim1=rows_left,
//           tile_dim0=HDc, tile_dim1=rows, tensor_dim0_stride=HDc}.
// Wave-uniform args; tracked with TENSORcnt.
// ---------------------------------------------------------------------------
__device__ __forceinline__ void tdm_load_tile(unsigned lds_addr, const bf16* g,
                                              int rows, int rows_left) {
#if USE_TDM
  typedef unsigned int u32x4 __attribute__((ext_vector_type(4)));
  typedef int          i32x4 __attribute__((ext_vector_type(4)));
  typedef int          i32x8 __attribute__((ext_vector_type(8)));
  const unsigned long long ga = (unsigned long long)(size_t)g;
  u32x4 g0;
  g0[0] = 1u;                                   // count=1 user descriptor
  g0[1] = lds_addr;                             // LDS byte address
  g0[2] = (unsigned)ga;                         // global_addr[31:0]
  g0[3] = (unsigned)((ga >> 32) & 0x01FFFFFFu)  // global_addr[56:32]
          | (2u << 30);                         // type = 2 ("image")
  i32x8 g1;
  g1[0] = (1 << 16);                            // data_size = 2 bytes
  g1[1] = (int)((unsigned)HDc << 16);           // tensor_dim0[15:0] @ [31:16]
  g1[2] = (int)((unsigned)rows_left << 16);     // dim0 hi=0 | tensor_dim1 lo16
  g1[3] = (int)((unsigned)HDc << 16);           // dim1 hi=0 | tile_dim0
  g1[4] = rows;                                 // tile_dim1 | tile_dim2=0
  g1[5] = HDc;                                  // tensor_dim0_stride[31:0]
  g1[6] = 0;                                    // stride0 hi | stride1 lo
  g1[7] = 0;                                    // stride1 hi
  i32x4 z4;
#pragma unroll
  for (int i = 0; i < 4; ++i) z4[i] = 0;
#if defined(__clang_major__) && (__clang_major__ >= 23)
  i32x8 z8;
#pragma unroll
  for (int i = 0; i < 8; ++i) z8[i] = 0;
  __builtin_amdgcn_tensor_load_to_lds(g0, g1, z4, z4, z8, 0);
#else
  __builtin_amdgcn_tensor_load_to_lds(g0, g1, z4, z4, 0);
#endif
#else
  (void)lds_addr; (void)g; (void)rows; (void)rows_left;
#endif
}

// ---------------------------------------------------------------------------
// Stage 0: f32 -> bf16 conversion (grid-stride, float4 loads).
// ---------------------------------------------------------------------------
__global__ __launch_bounds__(256) void k_cvt_bf16(const float* __restrict__ s,
                                                  bf16* __restrict__ d, int n4) {
  const float4* s4 = (const float4*)s;
  int i = blockIdx.x * 256 + threadIdx.x;
  const int stride = gridDim.x * 256;
  for (; i < n4; i += stride) {
    const float4 v = s4[i];
    d[4 * i + 0] = (bf16)v.x;
    d[4 * i + 1] = (bf16)v.y;
    d[4 * i + 2] = (bf16)v.z;
    d[4 * i + 3] = (bf16)v.w;
  }
}

// ---------------------------------------------------------------------------
// Stage 1: fused QKV GEMM + GPT-J RoPE epilogue.
// out[m,n] = sum_k X[m,k] * W[n,k]   (einsum 'bsh,oh->bso')
// Block: 256 thr = 8 waves (2 M x 4 N), block tile 64x128, wave tile 32x32.
// grid = (M/64, N/128, 3) — z selects {Q,K,V}. Q pre-scaled by 1/sqrt(HD)=1/16.
// Results scattered as bf16 to [B, NH, S, HD].
// ---------------------------------------------------------------------------
__global__ __launch_bounds__(256) void k_qkv_rope(
    const bf16* __restrict__ X, const bf16* __restrict__ Wq,
    const bf16* __restrict__ Wk, const bf16* __restrict__ Wv,
    const int* __restrict__ pos, bf16* __restrict__ Qo, bf16* __restrict__ Ko,
    bf16* __restrict__ Vo) {
  const int lane = threadIdx.x & 31;
  const int wave = threadIdx.x >> 5;
  const int m0 = blockIdx.x * 64 + (wave >> 2) * 32;
  const int n0 = blockIdx.y * 128 + (wave & 3) * 32;
  const int z  = blockIdx.z;
  const bf16* W   = (z == 0) ? Wq : (z == 1) ? Wk : Wv;
  bf16*       Out = (z == 0) ? Qo : (z == 1) ? Ko : Vo;

  f32x8 acc[2][2];
#pragma unroll
  for (int a = 0; a < 2; ++a)
#pragma unroll
    for (int b = 0; b < 2; ++b) acc[a][b] = zero8();

  for (int k0 = 0; k0 < Hc; k0 += 32) {
    const bf16x16 a0 = load_a_frag(X, Hc, m0, k0, lane);
    const bf16x16 a1 = load_a_frag(X, Hc, m0 + 16, k0, lane);
    const bf16x16 b0 = load_bt_frag(W, Hc, k0, n0, lane);
    const bf16x16 b1 = load_bt_frag(W, Hc, k0, n0 + 16, lane);
    acc[0][0] = wmma_bf16(a0, b0, acc[0][0]);
    acc[0][1] = wmma_bf16(a0, b1, acc[0][1]);
    acc[1][0] = wmma_bf16(a1, b0, acc[1][0]);
    acc[1][1] = wmma_bf16(a1, b1, acc[1][1]);
    // gfx1250 global_prefetch_b8 path for the next K slab
    __builtin_prefetch(X + (size_t)(m0 + (lane & 15)) * Hc + k0 + 64, 0, 1);
    __builtin_prefetch(W + (size_t)(n0 + (lane & 15)) * Hc + k0 + 64, 0, 1);
  }

  const int half = lane >> 4;
#pragma unroll
  for (int mt = 0; mt < 2; ++mt) {
#pragma unroll
    for (int nt = 0; nt < 2; ++nt) {
      f32x8 c = acc[mt][nt];
      const int nf = n0 + nt * 16 + (lane & 15);  // output feature o
      const int hh = nf >> 8;                     // head
      const int d  = nf & (HDc - 1);              // dim within head
      // d < RD is uniform across the wave (16-aligned fragment columns)
      const bool dorope = (z != 2) && (d < RDc);
      float fr = 0.0f;
      if (dorope) fr = __powf(10000.0f, -(float)(d & ~1) / (float)RDc);
#pragma unroll
      for (int i = 0; i < 8; ++i) {
        const int m  = m0 + mt * 16 + i + 8 * half;  // row = b*S + s
        const int bb = m >> 11;
        const int ss = m & (Sc - 1);
        float x = c[i];
        if (dorope) {
          // pair (2i, 2i+1) lives in lanes L, L^1 (same row)
          const float other = __shfl_xor(x, 1, 32);
          const float p = (float)pos[m];
          float sv, cv;
          __sincosf(p * fr, &sv, &cv);
          x = (d & 1) ? fmaf(other, sv, x * cv) : fmaf(x, cv, -(other * sv));
        }
        if (z == 0) x *= 0.0625f;  // fold 1/sqrt(256) into Q
        Out[((size_t)(bb * NHc + hh) * Sc + ss) * HDc + d] = (bf16)x;
      }
    }
  }
}

// ---------------------------------------------------------------------------
// Stage 2: causal flash attention. Block = 8 waves; each wave owns a 16-row Q
// tile. K/V chunks (32x256 bf16) staged in LDS by the TDM (wave 0 programs the
// descriptor, s_wait_tensorcnt 0, __syncthreads publishes) — cooperative b128
// loads as fallback. Probs go C->A through a per-wave LDS tile (same-wave DS
// ops are in-order, ISA 7.3). grid = (B*NH, S/128). Ctx bf16 as [B, S, H].
// ---------------------------------------------------------------------------
__global__ __launch_bounds__(256) void k_attn(const bf16* __restrict__ Q,
                                              const bf16* __restrict__ K,
                                              const bf16* __restrict__ V,
                                              bf16* __restrict__ Ctx) {
  __shared__ bf16 lk[32 * HDc];      // 16 KB
  __shared__ bf16 lv[32 * HDc];      // 16 KB
  __shared__ bf16 lp[8][16 * 32];    // 8 KB (per-wave probs staging)

  const int lane = threadIdx.x & 31;
  const int wave = threadIdx.x >> 5;
  const int half = lane >> 4;
  const int bh = blockIdx.x;
  const int bb = bh >> 4;
  const int hh = bh & (NHc - 1);
  const int qb = blockIdx.y * 128;
  const int q0 = qb + wave * 16;

  const bf16* Qb = Q + ((size_t)bh * Sc + q0) * HDc;
  const bf16* Kb = K + (size_t)bh * Sc * HDc;
  const bf16* Vb = V + (size_t)bh * Sc * HDc;

  bf16x16 aq[8];  // Q tile, K-dim = HD = 8 x 32
#pragma unroll
  for (int t = 0; t < 8; ++t) aq[t] = load_a_frag(Qb, HDc, 0, 32 * t, lane);

  f32x8 ctx[16];
#pragma unroll
  for (int n = 0; n < 16; ++n) ctx[n] = zero8();
  float mrun[8], lrun[8];
#pragma unroll
  for (int i = 0; i < 8; ++i) { mrun[i] = -3.0e38f; lrun[i] = 0.0f; }

  const int kend = qb + 128;  // block-uniform trip count (<= S)
  for (int kc = 0; kc < kend; kc += 32) {
#if USE_TDM
    if (wave == 0) {  // one wave programs the DMA; TENSORcnt tracks it
      tdm_load_tile((unsigned)(size_t)&lk[0], Kb + (size_t)kc * HDc, 32,
                    Sc - kc);
      tdm_load_tile((unsigned)(size_t)&lv[0], Vb + (size_t)kc * HDc, 32,
                    Sc - kc);
      __builtin_amdgcn_s_wait_tensorcnt(0);
    }
#else
    {  // cooperative 16KB+16KB chunk load: 1024 uint4 each, 256 threads x 4
      const uint4* gk = (const uint4*)(Kb + (size_t)kc * HDc);
      const uint4* gv = (const uint4*)(Vb + (size_t)kc * HDc);
      uint4* sk = (uint4*)lk;
      uint4* sv = (uint4*)lv;
#pragma unroll
      for (int j = 0; j < 4; ++j) {
        const int idx = (int)threadIdx.x + 256 * j;
        sk[idx] = gk[idx];
        sv[idx] = gv[idx];
      }
    }
#endif
    __syncthreads();

    if (kc <= q0 + 15) {  // wave-uniform: skip fully masked chunks
      // scores S[16 x 32] = Q . K^T  (K-major layout => bt fragments)
      f32x8 c0 = zero8(), c1 = zero8();
#pragma unroll
      for (int t = 0; t < 8; ++t) {
        c0 = wmma_bf16(aq[t], load_bt_frag(lk, HDc, 32 * t, 0, lane), c0);
        c1 = wmma_bf16(aq[t], load_bt_frag(lk, HDc, 32 * t, 16, lane), c1);
      }
      // causal mask + per-row chunk max
      const int col0 = kc + (lane & 15);
      const int col1 = col0 + 16;
      float cm[8];
#pragma unroll
      for (int i = 0; i < 8; ++i) {
        const int qr = q0 + i + 8 * half;
        const float s0 = (col0 <= qr) ? c0[i] : -3.0e38f;
        const float s1 = (col1 <= qr) ? c1[i] : -3.0e38f;
        c0[i] = s0;
        c1[i] = s1;
        cm[i] = fmaxf(s0, s1);
      }
#pragma unroll
      for (int off = 1; off < 16; off <<= 1)
#pragma unroll
        for (int i = 0; i < 8; ++i)
          cm[i] = fmaxf(cm[i], __shfl_xor(cm[i], off, 32));
      // online softmax update
      float alpha[8], rs[8];
#pragma unroll
      for (int i = 0; i < 8; ++i) {
        const float mn = fmaxf(mrun[i], cm[i]);
        alpha[i] = __expf(mrun[i] - mn);
        mrun[i] = mn;
        const float p0 = __expf(c0[i] - mn);  // masked -> underflows to 0
        const float p1 = __expf(c1[i] - mn);
        rs[i] = p0 + p1;
        const int row = i + 8 * half;
        lp[wave][row * 32 + (lane & 15)]      = (bf16)p0;
        lp[wave][row * 32 + 16 + (lane & 15)] = (bf16)p1;
      }
#pragma unroll
      for (int off = 1; off < 16; off <<= 1)
#pragma unroll
        for (int i = 0; i < 8; ++i) rs[i] += __shfl_xor(rs[i], off, 32);
#pragma unroll
      for (int i = 0; i < 8; ++i) lrun[i] = lrun[i] * alpha[i] + rs[i];
#pragma unroll
      for (int n = 0; n < 16; ++n)
#pragma unroll
        for (int i = 0; i < 8; ++i) ctx[n][i] *= alpha[i];
      // probs C-layout -> A-layout through LDS (same-wave DS: in order)
      const bf16x16 ap = load_a_frag(&lp[wave][0], 32, 0, 0, lane);
#pragma unroll
      for (int n = 0; n < 16; ++n)
        ctx[n] = wmma_bf16(ap, load_b_frag(lv, HDc, 0, 16 * n, lane), ctx[n]);
    }
    __syncthreads();  // LDS consumed before next chunk's DMA/loads
  }

  float inv[8];
#pragma unroll
  for (int i = 0; i < 8; ++i) inv[i] = 1.0f / lrun[i];  // diag => lrun >= ~1
#pragma unroll
  for (int n = 0; n < 16; ++n) {
    const int d = 16 * n + (lane & 15);
#pragma unroll
    for (int i = 0; i < 8; ++i) {
      const int sRow = q0 + i + 8 * half;
      Ctx[(size_t)(bb * Sc + sRow) * Hc + hh * HDc + d] =
          (bf16)(ctx[n][i] * inv[i]);
    }
  }
}

// ---------------------------------------------------------------------------
// Stage 3: output projection, f32 result. Same tiling as stage 1.
// ---------------------------------------------------------------------------
__global__ __launch_bounds__(256) void k_out_gemm(const bf16* __restrict__ X,
                                                  const bf16* __restrict__ Wo,
                                                  float* __restrict__ Out) {
  const int lane = threadIdx.x & 31;
  const int wave = threadIdx.x >> 5;
  const int m0 = blockIdx.x * 64 + (wave >> 2) * 32;
  const int n0 = blockIdx.y * 128 + (wave & 3) * 32;

  f32x8 acc[2][2];
#pragma unroll
  for (int a = 0; a < 2; ++a)
#pragma unroll
    for (int b = 0; b < 2; ++b) acc[a][b] = zero8();

  for (int k0 = 0; k0 < Hc; k0 += 32) {
    const bf16x16 a0 = load_a_frag(X, Hc, m0, k0, lane);
    const bf16x16 a1 = load_a_frag(X, Hc, m0 + 16, k0, lane);
    const bf16x16 b0 = load_bt_frag(Wo, Hc, k0, n0, lane);
    const bf16x16 b1 = load_bt_frag(Wo, Hc, k0, n0 + 16, lane);
    acc[0][0] = wmma_bf16(a0, b0, acc[0][0]);
    acc[0][1] = wmma_bf16(a0, b1, acc[0][1]);
    acc[1][0] = wmma_bf16(a1, b0, acc[1][0]);
    acc[1][1] = wmma_bf16(a1, b1, acc[1][1]);
    __builtin_prefetch(X + (size_t)(m0 + (lane & 15)) * Hc + k0 + 64, 0, 1);
    __builtin_prefetch(Wo + (size_t)(n0 + (lane & 15)) * Hc + k0 + 64, 0, 1);
  }

  const int half = lane >> 4;
#pragma unroll
  for (int mt = 0; mt < 2; ++mt)
#pragma unroll
    for (int nt = 0; nt < 2; ++nt)
#pragma unroll
      for (int i = 0; i < 8; ++i)
        Out[(size_t)(m0 + mt * 16 + i + 8 * half) * Hc + n0 + nt * 16 +
            (lane & 15)] = acc[mt][nt][i];
}

// ---------------------------------------------------------------------------
extern "C" void kernel_launch(void* const* d_in, const int* in_sizes, int n_in,
                              void* d_out, int out_size, void* d_ws,
                              size_t ws_size, hipStream_t stream) {
  const int*   pos = (const int*)d_in[0];
  const float* hid = (const float*)d_in[1];
  const float* wq  = (const float*)d_in[2];
  const float* wk  = (const float*)d_in[3];
  const float* wv  = (const float*)d_in[4];
  const float* wo  = (const float*)d_in[5];
  float* out = (float*)d_out;

  constexpr size_t E = (size_t)4096 * 4096;  // 16M elems per bf16 buffer
  char* w = (char*)d_ws;                     // needs 9 * 32MB = 288MB
  bf16* Xb   = (bf16*)(w + 0 * E * 2);
  bf16* Wqb  = (bf16*)(w + 1 * E * 2);
  bf16* Wkb  = (bf16*)(w + 2 * E * 2);
  bf16* Wvb  = (bf16*)(w + 3 * E * 2);
  bf16* Wob  = (bf16*)(w + 4 * E * 2);
  bf16* Qb   = (bf16*)(w + 5 * E * 2);
  bf16* Kb   = (bf16*)(w + 6 * E * 2);
  bf16* Vb   = (bf16*)(w + 7 * E * 2);
  bf16* Ctxb = (bf16*)(w + 8 * E * 2);

  const int n4 = (int)(E / 4);
  k_cvt_bf16<<<4096, 256, 0, stream>>>(hid, Xb, n4);
  k_cvt_bf16<<<4096, 256, 0, stream>>>(wq, Wqb, n4);
  k_cvt_bf16<<<4096, 256, 0, stream>>>(wk, Wkb, n4);
  k_cvt_bf16<<<4096, 256, 0, stream>>>(wv, Wvb, n4);
  k_cvt_bf16<<<4096, 256, 0, stream>>>(wo, Wob, n4);

  k_qkv_rope<<<dim3(64, 32, 3), 256, 0, stream>>>(Xb, Wqb, Wkb, Wvb, pos, Qb,
                                                  Kb, Vb);
  k_attn<<<dim3(Bc * NHc, Sc / 128), 256, 0, stream>>>(Qb, Kb, Vb, Ctxb);
  k_out_gemm<<<dim3(64, 32), 256, 0, stream>>>(Ctxb, Wob, out);
}
